// EncoderLayer_36593121362114
// MI455X (gfx1250) — compile-verified, matrix operand order checked by reading, not run on hardware
//
#include <hip/hip_runtime.h>
#include <hip/hip_bf16.h>

// GNN encoder layer for MI455X (gfx1250), wave32 + WMMA f32_16x16x32_f16.
// N=50000, K=16, H=128, E=64. Block = 16 nodes (256 edge rows), 256 threads
// (8 waves); each wave owns TWO 16-row M-tiles so every B-fragment load
// feeds two WMMAs (halves weight-side L2 traffic vs 1 tile/wave).

typedef __attribute__((ext_vector_type(16))) _Float16 v16h;
typedef __attribute__((ext_vector_type(8)))  _Float16 v8h;
typedef __attribute__((ext_vector_type(8)))  float    v8f;

#define NNODES 50000
#define KNBR   16
#define HDIM   128
#define EDIM   64

__device__ __forceinline__ v8f v8f_zero() {
  v8f z;
#pragma unroll
  for (int i = 0; i < 8; ++i) z[i] = 0.0f;
  return z;
}

__device__ __forceinline__ v8f wmma16(v16h a, v16h b, v8f c) {
  // emits v_wmma_f32_16x16x32_f16
  return __builtin_amdgcn_wmma_f32_16x16x32_f16(false, a, false, b, (short)0, c,
                                                false, false);
}

// A-fragment (16x32 f16). Lane L: r=L&15, hi=L>>4.
// halves 0..7  = A[r][c0 + hi*8 .. +7]      (contiguous 16B)
// halves 8..15 = A[r][c0 + 16 + hi*8 .. +7] (contiguous 16B)
__device__ __forceinline__ v16h load_a_frag(const _Float16* base, int stride, int c0) {
  const int lane = threadIdx.x & 31;
  const int r = lane & 15;
  const int hi = lane >> 4;
  const _Float16* p = base + r * stride + c0 + hi * 8;
  v8h lo = *(const v8h*)(p);
  v8h up = *(const v8h*)(p + 16);
  v16h out;
#pragma unroll
  for (int i = 0; i < 8; ++i) { out[i] = lo[i]; out[i + 8] = up[i]; }
  return out;
}

// B-fragment prepacked in ws: [tile][lane][16 halves] (32B per lane, contiguous)
__device__ __forceinline__ v16h load_b_frag(const _Float16* fragbase) {
  const int lane = threadIdx.x & 31;
  return *(const v16h*)(fragbase + lane * 16);
}

// ---- prep: pack a KDIMxNDIM f32 weight into WMMA-B f16 fragment layout ----
__global__ void pack_b_frags(const float* __restrict__ W, _Float16* __restrict__ out,
                             int KDIM, int NDIM) {
  int o = blockIdx.x * blockDim.x + threadIdx.x;
  if (o >= KDIM * NDIM) return;
  int ntn  = NDIM >> 4;
  int m    = o & 15;
  int L    = (o >> 4) & 31;
  int tile = o >> 9;
  int tn   = tile % ntn;
  int tk   = tile / ntn;
  int c    = L & 15;
  int hi   = L >> 4;
  int krel = (m < 8) ? (hi * 8 + m) : (16 + hi * 8 + (m - 8));
  out[o] = (_Float16)W[(tk * 32 + krel) * NDIM + tn * 16 + c];
}

// =======================  Kernel 1: messages + node update  =======================
__global__ __launch_bounds__(256) void msg_node_kernel(
    const float* __restrict__ nodes, const float* __restrict__ edges,
    const int* __restrict__ knn,
    const _Float16* __restrict__ mW1f, const float* __restrict__ mb1,
    const _Float16* __restrict__ mW2f, const float* __restrict__ mb2,
    const _Float16* __restrict__ nW1f, const float* __restrict__ nb1,
    const _Float16* __restrict__ nW2f, const float* __restrict__ nb2,
    const float* __restrict__ ln_ng, const float* __restrict__ ln_nb,
    float* __restrict__ out_nodes)
{
  extern __shared__ char smem_raw[];
  _Float16* srcb  = (_Float16*)smem_raw;       // 16x128       (this block's nodes, f16)
  _Float16* edgeb = srcb  + 16 * 128;          // 256x64       (edge features, f16)
  _Float16* nbrb  = edgeb + 256 * 64;          // 256x128      (gathered nbrs; reused as msg-hidden)
  _Float16* aggh  = nbrb  + 256 * 128;         // 16x128       (aggregated messages, f16)
  _Float16* nodeh = aggh  + 16 * 128;          // 16x128       (node MLP hidden, f16)
  float*    preln = (float*)(nodeh + 16 * 128);// 16x128 f32   (pre-LayerNorm)

  const int node0 = blockIdx.x * 16;
  const int tid   = threadIdx.x;

  // ---- stage activations to LDS as f16 ----
  for (int i = tid; i < 16 * 128; i += 256)
    srcb[i] = (_Float16)nodes[(size_t)node0 * 128 + i];

  {
    const float4* e4 = (const float4*)(edges + (size_t)blockIdx.x * 256 * 64);
    for (int i = tid; i < 256 * 16; i += 256) {
      float4 v = e4[i];
      _Float16* d = edgeb + i * 4;
      d[0] = (_Float16)v.x; d[1] = (_Float16)v.y;
      d[2] = (_Float16)v.z; d[3] = (_Float16)v.w;
    }
  }
  for (int i = tid; i < 256 * 32; i += 256) {      // 256 rows x 32 float4
    int p = i >> 5, cq = i & 31;
    int idx = knn[node0 * 16 + p];
    float4 v = ((const float4*)(nodes + (size_t)idx * 128))[cq];
    _Float16* d = nbrb + p * 128 + cq * 4;
    d[0] = (_Float16)v.x; d[1] = (_Float16)v.y;
    d[2] = (_Float16)v.z; d[3] = (_Float16)v.w;
  }
  __syncthreads();

  const int wave = tid >> 5;      // 0..7 : owns M-tiles 2w and 2w+1 (= nodes 2w, 2w+1)
  const int lane = tid & 31;
  const int c    = lane & 15;
  const int hi   = lane >> 4;
  const int t0   = wave * 2;

  // ---- message MLP layer 1: (256x320) @ (320x128), 2 M-tiles per wave ----
  v8f acc[2][8];
#pragma unroll
  for (int t = 0; t < 2; ++t)
#pragma unroll
    for (int n = 0; n < 8; ++n) acc[t][n] = v8f_zero();

  for (int kk = 0; kk < 10; ++kk) {
    v16h a[2];
#pragma unroll
    for (int t = 0; t < 2; ++t) {
      const int mt = t0 + t;
      const _Float16* ab; int as, c0;
      if (kk < 4)      { ab = srcb  + mt * 128;      as = 0;   c0 = kk * 32; }       // src (rows identical)
      else if (kk < 8) { ab = nbrb  + mt * 16 * 128; as = 128; c0 = (kk - 4) * 32; } // nbr
      else             { ab = edgeb + mt * 16 * 64;  as = 64;  c0 = (kk - 8) * 32; } // edge
      a[t] = load_a_frag(ab, as, c0);
    }
#pragma unroll
    for (int nn = 0; nn < 8; ++nn) {
      v16h b = load_b_frag(mW1f + (kk * 8 + nn) * 512);   // loaded once, used twice
      acc[0][nn] = wmma16(a[0], b, acc[0][nn]);
      acc[1][nn] = wmma16(a[1], b, acc[1][nn]);
    }
  }

  // relu + bias -> f16 hidden, overlaying this wave's own nbr rows (no cross-wave hazard)
#pragma unroll
  for (int t = 0; t < 2; ++t) {
    _Float16* hb = nbrb + (t0 + t) * 16 * 128;
#pragma unroll
    for (int nn = 0; nn < 8; ++nn) {
      float bv = mb1[nn * 16 + c];
#pragma unroll
      for (int j = 0; j < 8; ++j) {
        float v = acc[t][nn][j] + bv;
        v = v > 0.0f ? v : 0.0f;
        hb[(hi * 8 + j) * 128 + nn * 16 + c] = (_Float16)v;
      }
    }
  }

  // ---- message MLP layer 2: (256x128) @ (128x128) ----
  v8f acc2[2][8];
#pragma unroll
  for (int t = 0; t < 2; ++t)
#pragma unroll
    for (int n = 0; n < 8; ++n) acc2[t][n] = v8f_zero();
  for (int kk = 0; kk < 4; ++kk) {
    v16h a[2];
#pragma unroll
    for (int t = 0; t < 2; ++t)
      a[t] = load_a_frag(nbrb + (t0 + t) * 16 * 128, 128, kk * 32);
#pragma unroll
    for (int nn = 0; nn < 8; ++nn) {
      v16h b = load_b_frag(mW2f + (kk * 8 + nn) * 512);
      acc2[0][nn] = wmma16(a[0], b, acc2[0][nn]);
      acc2[1][nn] = wmma16(a[1], b, acc2[1][nn]);
    }
  }

  // ---- aggregate over k (sum 16 rows of each C tile) ----
#pragma unroll
  for (int t = 0; t < 2; ++t) {
#pragma unroll
    for (int nn = 0; nn < 8; ++nn) {
      float s = 0.0f;
#pragma unroll
      for (int j = 0; j < 8; ++j) s += acc2[t][nn][j];
      s += __shfl_xor(s, 16, 32);          // combine hi halves (rows 0-7 + 8-15)
      s += 16.0f * mb2[nn * 16 + c];       // bias added once per row, 16 rows
      if (hi == 0) aggh[(t0 + t) * 128 + nn * 16 + c] = (_Float16)s;
    }
  }
  __syncthreads();

  // ---- node MLP: [nodes | agg] (16x256) -> 128 -> relu -> 128; wave = N-tile ----
  {
    v8f na = v8f_zero();
    for (int kk = 0; kk < 8; ++kk) {
      const _Float16* ab = (kk < 4) ? srcb : aggh;
      v16h a = load_a_frag(ab, 128, (kk & 3) * 32);
      v16h b = load_b_frag(nW1f + (kk * 8 + wave) * 512);
      na = wmma16(a, b, na);
    }
    float bv = nb1[wave * 16 + c];
#pragma unroll
    for (int j = 0; j < 8; ++j) {
      float v = na[j] + bv;
      v = v > 0.0f ? v : 0.0f;
      nodeh[(hi * 8 + j) * 128 + wave * 16 + c] = (_Float16)v;
    }
  }
  __syncthreads();
  {
    v8f n2 = v8f_zero();
    for (int kk = 0; kk < 4; ++kk) {
      v16h a = load_a_frag(nodeh, 128, kk * 32);
      v16h b = load_b_frag(nW2f + (kk * 8 + wave) * 512);
      n2 = wmma16(a, b, n2);
    }
    float bv = nb2[wave * 16 + c];
#pragma unroll
    for (int j = 0; j < 8; ++j) {
      int r = hi * 8 + j;
      float res = nodes[(size_t)(node0 + r) * 128 + wave * 16 + c];
      preln[r * 128 + wave * 16 + c] = n2[j] + bv + res;
    }
  }
  __syncthreads();

  // ---- LayerNorm over H=128: wave w handles rows w and w+8 (4 elems/lane) ----
  for (int r = wave; r < 16; r += 8) {
    const float* pr = preln + r * 128 + lane * 4;
    float x0 = pr[0], x1 = pr[1], x2 = pr[2], x3 = pr[3];
    float s  = x0 + x1 + x2 + x3;
    float ss = x0 * x0 + x1 * x1 + x2 * x2 + x3 * x3;
#pragma unroll
    for (int off = 1; off < 32; off <<= 1) {
      s  += __shfl_xor(s,  off, 32);
      ss += __shfl_xor(ss, off, 32);
    }
    float mu  = s * (1.0f / 128.0f);
    float var = ss * (1.0f / 128.0f) - mu * mu;
    float inv = rsqrtf(var + 1e-5f);
    float* o = out_nodes + (size_t)(node0 + r) * 128 + lane * 4;
    const int cb = lane * 4;
    o[0] = (x0 - mu) * inv * ln_ng[cb + 0] + ln_nb[cb + 0];
    o[1] = (x1 - mu) * inv * ln_ng[cb + 1] + ln_nb[cb + 1];
    o[2] = (x2 - mu) * inv * ln_ng[cb + 2] + ln_nb[cb + 2];
    o[3] = (x3 - mu) * inv * ln_ng[cb + 3] + ln_nb[cb + 3];
  }
}

// =======================  Kernel 2: edge update (uses updated nodes)  ==============
__global__ __launch_bounds__(256) void edge_kernel(
    const float* __restrict__ unodes, const float* __restrict__ edges,
    const int* __restrict__ knn,
    const _Float16* __restrict__ eW1f, const float* __restrict__ eb1,
    const _Float16* __restrict__ eW2f, const float* __restrict__ eb2,
    const float* __restrict__ ln_eg, const float* __restrict__ ln_eb,
    float* __restrict__ out_edges)
{
  extern __shared__ char smem_raw[];
  _Float16* srcb  = (_Float16*)smem_raw;   // 16x128 (updated nodes of this block)
  _Float16* edgeb = srcb  + 16 * 128;      // 256x64
  _Float16* nbrb  = edgeb + 256 * 64;      // 256x128 (gathered updated nbrs; reused as hidden)

  const int node0 = blockIdx.x * 16;
  const int tid   = threadIdx.x;

  for (int i = tid; i < 16 * 128; i += 256)
    srcb[i] = (_Float16)unodes[(size_t)node0 * 128 + i];
  {
    const float4* e4 = (const float4*)(edges + (size_t)blockIdx.x * 256 * 64);
    for (int i = tid; i < 256 * 16; i += 256) {
      float4 v = e4[i];
      _Float16* d = edgeb + i * 4;
      d[0] = (_Float16)v.x; d[1] = (_Float16)v.y;
      d[2] = (_Float16)v.z; d[3] = (_Float16)v.w;
    }
  }
  for (int i = tid; i < 256 * 32; i += 256) {
    int p = i >> 5, cq = i & 31;
    int idx = knn[node0 * 16 + p];
    float4 v = ((const float4*)(unodes + (size_t)idx * 128))[cq];
    _Float16* d = nbrb + p * 128 + cq * 4;
    d[0] = (_Float16)v.x; d[1] = (_Float16)v.y;
    d[2] = (_Float16)v.z; d[3] = (_Float16)v.w;
  }
  __syncthreads();

  const int wave = tid >> 5;   // 0..7, owns M-tiles 2w, 2w+1
  const int lane = tid & 31;
  const int c    = lane & 15;
  const int hi   = lane >> 4;
  const int t0   = wave * 2;

  // ---- edge MLP layer 1: (256x320) @ (320x64), 2 M-tiles per wave ----
  v8f acc[2][4];
#pragma unroll
  for (int t = 0; t < 2; ++t)
#pragma unroll
    for (int n = 0; n < 4; ++n) acc[t][n] = v8f_zero();
  for (int kk = 0; kk < 10; ++kk) {
    v16h a[2];
#pragma unroll
    for (int t = 0; t < 2; ++t) {
      const int mt = t0 + t;
      const _Float16* ab; int as, c0;
      if (kk < 4)      { ab = srcb  + mt * 128;      as = 0;   c0 = kk * 32; }
      else if (kk < 8) { ab = nbrb  + mt * 16 * 128; as = 128; c0 = (kk - 4) * 32; }
      else             { ab = edgeb + mt * 16 * 64;  as = 64;  c0 = (kk - 8) * 32; }
      a[t] = load_a_frag(ab, as, c0);
    }
#pragma unroll
    for (int nn = 0; nn < 4; ++nn) {
      v16h b = load_b_frag(eW1f + (kk * 4 + nn) * 512);
      acc[0][nn] = wmma16(a[0], b, acc[0][nn]);
      acc[1][nn] = wmma16(a[1], b, acc[1][nn]);
    }
  }

  // relu + bias -> f16 hidden (overlay own nbr rows, stride 64)
#pragma unroll
  for (int t = 0; t < 2; ++t) {
    _Float16* hb = nbrb + (t0 + t) * 16 * 128;
#pragma unroll
    for (int nn = 0; nn < 4; ++nn) {
      float bv = eb1[nn * 16 + c];
#pragma unroll
      for (int j = 0; j < 8; ++j) {
        float v = acc[t][nn][j] + bv;
        v = v > 0.0f ? v : 0.0f;
        hb[(hi * 8 + j) * 64 + nn * 16 + c] = (_Float16)v;
      }
    }
  }

  // ---- edge MLP layer 2: (256x64) @ (64x64) ----
  v8f a2[2][4];
#pragma unroll
  for (int t = 0; t < 2; ++t)
#pragma unroll
    for (int n = 0; n < 4; ++n) a2[t][n] = v8f_zero();
  for (int kk = 0; kk < 2; ++kk) {
    v16h a[2];
#pragma unroll
    for (int t = 0; t < 2; ++t)
      a[t] = load_a_frag(nbrb + (t0 + t) * 16 * 128, 64, kk * 32);
#pragma unroll
    for (int nn = 0; nn < 4; ++nn) {
      v16h b = load_b_frag(eW2f + (kk * 4 + nn) * 512);
      a2[0][nn] = wmma16(a[0], b, a2[0][nn]);
      a2[1][nn] = wmma16(a[1], b, a2[1][nn]);
    }
  }

  // ---- residual + per-row LayerNorm over E=64, entirely in registers ----
#pragma unroll
  for (int t = 0; t < 2; ++t) {
    const int mt = t0 + t;
    float x[4][8];
#pragma unroll
    for (int nn = 0; nn < 4; ++nn) {
      float bv = eb2[nn * 16 + c];
#pragma unroll
      for (int j = 0; j < 8; ++j) {
        int r = mt * 16 + hi * 8 + j;
        size_t P = (size_t)blockIdx.x * 256 + r;
        x[nn][j] = a2[t][nn][j] + bv + edges[P * 64 + nn * 16 + c];
      }
    }
#pragma unroll
    for (int j = 0; j < 8; ++j) {
      float s  = x[0][j] + x[1][j] + x[2][j] + x[3][j];
      float ss = x[0][j] * x[0][j] + x[1][j] * x[1][j] +
                 x[2][j] * x[2][j] + x[3][j] * x[3][j];
#pragma unroll
      for (int off = 1; off < 16; off <<= 1) {   // reduce within 16-lane half (same rows)
        s  += __shfl_xor(s,  off, 32);
        ss += __shfl_xor(ss, off, 32);
      }
      float mu  = s * (1.0f / 64.0f);
      float var = ss * (1.0f / 64.0f) - mu * mu;
      float inv = rsqrtf(var + 1e-5f);
      int r = mt * 16 + hi * 8 + j;
      size_t P = (size_t)blockIdx.x * 256 + r;
#pragma unroll
      for (int nn = 0; nn < 4; ++nn) {
        int col = nn * 16 + c;
        out_edges[P * 64 + col] = (x[nn][j] - mu) * inv * ln_eg[col] + ln_eb[col];
      }
    }
  }
}

extern "C" void kernel_launch(void* const* d_in, const int* in_sizes, int n_in,
                              void* d_out, int out_size, void* d_ws, size_t ws_size,
                              hipStream_t stream) {
  const float* nodes = (const float*)d_in[0];
  const float* edges = (const float*)d_in[1];
  const int*   knn   = (const int*)d_in[2];
  const float* mW1 = (const float*)d_in[3];  const float* mb1 = (const float*)d_in[4];
  const float* mW2 = (const float*)d_in[5];  const float* mb2 = (const float*)d_in[6];
  const float* nW1 = (const float*)d_in[7];  const float* nb1 = (const float*)d_in[8];
  const float* nW2 = (const float*)d_in[9];  const float* nb2 = (const float*)d_in[10];
  const float* eW1 = (const float*)d_in[11]; const float* eb1 = (const float*)d_in[12];
  const float* eW2 = (const float*)d_in[13]; const float* eb2 = (const float*)d_in[14];
  const float* ln_ng = (const float*)d_in[15]; const float* ln_nb = (const float*)d_in[16];
  const float* ln_eg = (const float*)d_in[17]; const float* ln_eb = (const float*)d_in[18];

  // f16 B-fragment weight cache in workspace (halves): 131072 total = 256 KB
  _Float16* wsh  = (_Float16*)d_ws;
  _Float16* mW1f = wsh + 0;        // 320x128 -> 40960
  _Float16* mW2f = wsh + 40960;    // 128x128 -> 16384
  _Float16* nW1f = wsh + 57344;    // 256x128 -> 32768
  _Float16* nW2f = wsh + 90112;    // 128x128 -> 16384
  _Float16* eW1f = wsh + 106496;   // 320x64  -> 20480
  _Float16* eW2f = wsh + 126976;   // 64x64   ->  4096

  auto pack = [&](const float* W, _Float16* o, int K, int Nn) {
    int t = K * Nn;
    pack_b_frags<<<(t + 255) / 256, 256, 0, stream>>>(W, o, K, Nn);
  };
  pack(mW1, mW1f, 320, 128);
  pack(mW2, mW2f, 128, 128);
  pack(nW1, nW1f, 256, 128);
  pack(nW2, nW2f, 128, 128);
  pack(eW1, eW1f, 320, 64);
  pack(eW2, eW2f, 64, 64);

  float* out_nodes = (float*)d_out;
  float* out_edges = out_nodes + (size_t)NNODES * HDIM;

  const int blocks = NNODES / 16;  // 3125
  const size_t smem1 = (16 * 128 + 256 * 64 + 256 * 128 + 16 * 128 + 16 * 128) * sizeof(_Float16)
                     + 16 * 128 * sizeof(float);                       // 118784 B
  const size_t smem2 = (16 * 128 + 256 * 64 + 256 * 128) * sizeof(_Float16); // 102400 B

  msg_node_kernel<<<blocks, 256, smem1, stream>>>(
      nodes, edges, knn, mW1f, mb1, mW2f, mb2, nW1f, nb1, nW2f, nb2,
      ln_ng, ln_nb, out_nodes);
  edge_kernel<<<blocks, 256, smem2, stream>>>(
      out_nodes, edges, knn, eW1f, eb1, eW2f, eb2, ln_eg, ln_eb, out_edges);
}
